// Model_24799141167781
// MI455X (gfx1250) — compile-verified, matrix operand order checked by reading, not run on hardware
//
#include <hip/hip_runtime.h>
#include <hip/hip_bf16.h>

typedef __bf16 bf16_t;
typedef __attribute__((ext_vector_type(16))) __bf16 v16bf;
typedef __attribute__((ext_vector_type(8)))  __bf16 v8bf;
typedef __attribute__((ext_vector_type(8)))  float  v8f;

#define GAT_ALPHA 0.2f

// ---------------- elementwise / setup kernels ----------------

__global__ void fill_f32_kernel(float* __restrict__ p, float v, int n) {
  int i = blockIdx.x * blockDim.x + threadIdx.x;
  if (i < n) p[i] = v;
}

// tweet features: mean of 16 word embeddings per tweet, write bf16 (K padded)
// indices staged in LDS once per block (all 128 threads share the same 16)
__global__ void tweet_feat_kernel(const int* __restrict__ idx,
                                  const float* __restrict__ emb,
                                  bf16_t* __restrict__ X,
                                  int F, int Fp) {
  __shared__ int sidx[16];
  int t = blockIdx.x;
  if (threadIdx.x < 16) sidx[threadIdx.x] = idx[t * 16 + threadIdx.x];
  __syncthreads();
  for (int f = threadIdx.x; f < Fp; f += blockDim.x) {
    float s = 0.f;
    if (f < F) {
      #pragma unroll
      for (int w = 0; w < 16; ++w)
        s += emb[(size_t)sidx[w] * F + f];
      s *= (1.f / 16.f);
    }
    X[(size_t)t * Fp + f] = (bf16_t)s;
  }
}

// fp32 [N,F] -> bf16 [N,Fp] (zero pad)
__global__ void conv_x_kernel(const float* __restrict__ src, bf16_t* __restrict__ dst,
                              int N, int F, int Fp) {
  int i = blockIdx.x * blockDim.x + threadIdx.x;
  if (i >= N * Fp) return;
  int n = i / Fp, f = i - n * Fp;
  float v = (f < F) ? src[(size_t)n * F + f] : 0.f;
  dst[i] = (bf16_t)v;
}

// W [K,N] fp32 -> Wt [N,Kp] bf16 (transposed, zero pad K..Kp)
__global__ void conv_wt_kernel(const float* __restrict__ W, bf16_t* __restrict__ Wt,
                               int K, int N, int Kp) {
  int i = blockIdx.x * blockDim.x + threadIdx.x;
  if (i >= N * Kp) return;
  int n = i / Kp, k = i - n * Kp;
  float v = (k < K) ? W[(size_t)k * N + n] : 0.f;
  Wt[i] = (bf16_t)v;
}

// ---------------- WMMA GEMM: C[M,N] = A[M,K] * Bt[N,K]^T ----------------
// One wave computes a 16x64 strip (4 accumulators): A fragment loaded once per
// K-chunk, reused across 4 B tiles -> 2.5 b128 loads per WMMA instead of 4.
__device__ __forceinline__ v16bf load_frag(const bf16_t* p) {
  v8bf lo = *(const v8bf*)p;
  v8bf hi = *(const v8bf*)(p + 16);
  v16bf r;
  #pragma unroll
  for (int i = 0; i < 8; ++i) { r[i] = lo[i]; r[i + 8] = hi[i]; }
  return r;
}

__global__ void wmma_gemm_kernel(const bf16_t* __restrict__ A,
                                 const bf16_t* __restrict__ Bt,
                                 float* __restrict__ C,
                                 int M, int N, int K,
                                 int lda, int ldb, int ldc) {
  int wave = (blockIdx.x * blockDim.x + threadIdx.x) >> 5;
  int lane = threadIdx.x & 31;
  int ngrp = N >> 6;                     // groups of 64 output columns
  int tiles = (M >> 4) * ngrp;
  if (wave >= tiles) return;
  int tm = wave / ngrp;
  int m0 = tm << 4;
  int n0 = (wave - tm * ngrp) << 6;
  int half = lane >> 4;                  // 0: K-low group, 1: K-high group
  int lr   = lane & 15;                  // row (A) / col (B) within tile

  const bf16_t* arow = A  + (size_t)(m0 + lr) * lda + (half << 3);
  const bf16_t* brow = Bt + (size_t)(n0 + lr) * ldb + (half << 3);
  size_t bstep = (size_t)16 * ldb;

  v8f acc0 = {}, acc1 = {}, acc2 = {}, acc3 = {};
  for (int k0 = 0; k0 < K; k0 += 32) {
    v16bf a  = load_frag(arow + k0);
    v16bf b0 = load_frag(brow + k0);
    v16bf b1 = load_frag(brow + bstep + k0);
    v16bf b2 = load_frag(brow + 2 * bstep + k0);
    v16bf b3 = load_frag(brow + 3 * bstep + k0);
    acc0 = __builtin_amdgcn_wmma_f32_16x16x32_bf16(false, a, false, b0, (short)0, acc0, false, false);
    acc1 = __builtin_amdgcn_wmma_f32_16x16x32_bf16(false, a, false, b1, (short)0, acc1, false, false);
    acc2 = __builtin_amdgcn_wmma_f32_16x16x32_bf16(false, a, false, b2, (short)0, acc2, false, false);
    acc3 = __builtin_amdgcn_wmma_f32_16x16x32_bf16(false, a, false, b3, (short)0, acc3, false, false);
  }
  // C fragment: lane lr = column, VGPR r = row (half selects rows 0-7 / 8-15)
  float* cp = C + (size_t)(m0 + (half << 3)) * ldc + n0 + lr;
  #pragma unroll
  for (int r = 0; r < 8; ++r) {
    cp[(size_t)r * ldc +  0] = acc0[r];
    cp[(size_t)r * ldc + 16] = acc1[r];
    cp[(size_t)r * ldc + 32] = acc2[r];
    cp[(size_t)r * ldc + 48] = acc3[r];
  }
}

// ---------------- GAT attention kernels ----------------

// wave-per-node: f_src[n] = h[n]·a[0:d], f_dst[n] = h[n]·a[d:2d]  (coalesced)
__global__ void node_att_kernel(const float* __restrict__ h, const float* __restrict__ a,
                                float* __restrict__ fsrc, float* __restrict__ fdst,
                                int N, int d) {
  int wave = (blockIdx.x * blockDim.x + threadIdx.x) >> 5;
  int lane = threadIdx.x & 31;
  if (wave >= N) return;
  float s = 0.f, t = 0.f;
  for (int j = lane; j < d; j += 32) {
    float v = h[(size_t)wave * d + j];
    s += v * a[j];
    t += v * a[d + j];
  }
  #pragma unroll
  for (int o = 16; o > 0; o >>= 1) {
    s += __shfl_down(s, o);
    t += __shfl_down(t, o);
  }
  if (lane == 0) { fsrc[wave] = s; fdst[wave] = t; }
}

// e = exp(-leakyrelu(f_src[row]+f_dst[col])); denom[row] += e
__global__ void edge_att_kernel(const int* __restrict__ row, const int* __restrict__ col,
                                const float* __restrict__ fsrc, const float* __restrict__ fdst,
                                float* __restrict__ eatt, float* __restrict__ denom, int E) {
  int e = blockIdx.x * blockDim.x + threadIdx.x;
  if (e >= E) return;
  float x = fsrc[row[e]] + fdst[col[e]];
  float l = (x > 0.f) ? x : GAT_ALPHA * x;
  float v = __expf(-l);
  eatt[e] = v;
  atomicAdd(denom + row[e], v);
}

// num[row, j] += e * h[col, j]   (one thread per (edge, dim); f32 atomics in L2)
__global__ void edge_agg_kernel(const int* __restrict__ row, const int* __restrict__ col,
                                const float* __restrict__ eatt, const float* __restrict__ h,
                                float* __restrict__ num, long long total, int d, int dshift) {
  long long i = (long long)blockIdx.x * blockDim.x + threadIdx.x;
  if (i >= total) return;
  int e = (int)(i >> dshift);
  int j = (int)(i & (d - 1));
  atomicAdd(num + (size_t)row[e] * d + j, eatt[e] * h[(size_t)col[e] * d + j]);
}

// out = elu(num / (denom + eps)); write bf16 (always) and fp32 (optional)
__global__ void normalize_kernel(const float* __restrict__ num, const float* __restrict__ denom,
                                 float* __restrict__ outf, bf16_t* __restrict__ outbf,
                                 int total, int dshift, int write_f32) {
  int i = blockIdx.x * blockDim.x + threadIdx.x;
  if (i >= total) return;
  int n = i >> dshift;
  float v = num[i] / (denom[n] + 1e-16f);
  v = (v > 0.f) ? v : (__expf(v) - 1.f);   // ELU
  if (write_f32) outf[i] = v;
  outbf[i] = (bf16_t)v;
}

// ---------------- fusion / head kernels ----------------

// acc += sum_n sum_j tanh(U[n,j]) * proj[j]
// wave-per-node, LDS block reduction, one global atomic per 256-thread block
__global__ void att_reduce_kernel(const float* __restrict__ U, const float* __restrict__ proj,
                                  float* __restrict__ acc, int N, int d) {
  __shared__ float part[8];
  int wave = threadIdx.x >> 5, lane = threadIdx.x & 31;
  int n = blockIdx.x * (blockDim.x >> 5) + wave;
  float s = 0.f;
  if (n < N) {
    for (int j = lane; j < d; j += 32)
      s += tanhf(U[(size_t)n * d + j]) * proj[j];
  }
  #pragma unroll
  for (int o = 16; o > 0; o >>= 1) s += __shfl_down(s, o);
  if (lane == 0) part[wave] = s;
  __syncthreads();
  if (threadIdx.x == 0) {
    float t = 0.f;
    int nw = blockDim.x >> 5;
    for (int w = 0; w < nw; ++w) t += part[w];
    atomicAdd(acc, t);
  }
}

__global__ void att_softmax_kernel(const float* __restrict__ sums, float* __restrict__ att,
                                   float invN) {
  float a = sums[0] * invN, b = sums[1] * invN;
  float m = fmaxf(a, b);
  float ea = __expf(a - m), eb = __expf(b - m);
  float inv = 1.f / (ea + eb);
  att[0] = ea * inv;
  att[1] = eb * inv;
}

// wave-per-sample: feat = a0*twX[ti] + a1*tuX[ui]; logits; log_softmax
__global__ void final_out_kernel(const float* __restrict__ twX, const float* __restrict__ tuX,
                                 const int* __restrict__ ti, const int* __restrict__ ui,
                                 const float* __restrict__ att,
                                 const float* __restrict__ Wo, const float* __restrict__ bo,
                                 float* __restrict__ out, int B, int d) {
  int wave = (blockIdx.x * blockDim.x + threadIdx.x) >> 5;
  int lane = threadIdx.x & 31;
  if (wave >= B) return;
  float a0 = att[0], a1 = att[1];
  const float* x0 = twX + (size_t)ti[wave] * d;
  const float* x1 = tuX + (size_t)ui[wave] * d;
  float l0 = 0.f, l1 = 0.f;
  for (int j = lane; j < d; j += 32) {
    float f = a0 * x0[j] + a1 * x1[j];
    l0 += f * Wo[j];
    l1 += f * Wo[d + j];
  }
  #pragma unroll
  for (int o = 16; o > 0; o >>= 1) {
    l0 += __shfl_down(l0, o);
    l1 += __shfl_down(l1, o);
  }
  if (lane == 0) {
    l0 += bo[0]; l1 += bo[1];
    float m   = fmaxf(l0, l1);
    float lse = m + logf(__expf(l0 - m) + __expf(l1 - m));
    out[2 * wave]     = l0 - lse;
    out[2 * wave + 1] = l1 - lse;
  }
}

// ---------------- host orchestration ----------------

static inline int cdiv(long long a, long long b) { return (int)((a + b - 1) / b); }

extern "C" void kernel_launch(void* const* d_in, const int* in_sizes, int n_in,
                              void* d_out, int out_size, void* d_ws, size_t ws_size,
                              hipStream_t stream) {
  (void)in_sizes; (void)n_in; (void)out_size; (void)ws_size;

  const int kN  = 50000;   // nodes per graph (tweets / users)
  const int kF  = 300;     // input feature dim
  const int kFp = 320;     // padded to multiple of 32 for WMMA K
  const int kE  = 800000;  // edges per graph
  const int kD1 = 64;      // hidden dim
  const int kD2 = 128;     // joint dim
  const int kB  = 4096;    // batch

  // inputs in setup_inputs() dict order
  const int*   features_index = (const int*)d_in[0];
  const int*   tw_edges       = (const int*)d_in[1];
  const int*   ut_edges       = (const int*)d_in[2];
  const int*   tw_gidx        = (const int*)d_in[3];
  const int*   ut_gidx        = (const int*)d_in[4];
  const float* word_emb       = (const float*)d_in[5];
  const float* user_emb       = (const float*)d_in[6];
  const float* tw_W1 = (const float*)d_in[7];
  const float* tw_a1 = (const float*)d_in[8];
  const float* tw_W2 = (const float*)d_in[9];
  const float* tw_a2 = (const float*)d_in[10];
  const float* tu_W1 = (const float*)d_in[11];
  const float* tu_a1 = (const float*)d_in[12];
  const float* tu_W2 = (const float*)d_in[13];
  const float* tu_a2 = (const float*)d_in[14];
  const float* weight_W    = (const float*)d_in[15];
  const float* weight_proj = (const float*)d_in[16];
  const float* out_W       = (const float*)d_in[17];
  const float* out_b       = (const float*)d_in[18];
  float* out = (float*)d_out;

  // workspace carve-out (256B aligned chunks)
  char* base = (char*)d_ws;
  size_t off = 0;
  auto alloc = [&](size_t bytes) -> void* {
    void* p = base + off;
    off += (bytes + 255) & ~(size_t)255;
    return p;
  };
  bf16_t* Xbf   = (bf16_t*)alloc((size_t)kN * kFp * 2);   // shared input buffer (tw then tu)
  bf16_t* Wt    = (bf16_t*)alloc((size_t)kD1 * kFp * 2);  // largest transposed weight (64x320)
  float*  h     = (float*)alloc((size_t)kN * kD2 * 4);    // GEMM output / U buffer
  float*  fsrc  = (float*)alloc((size_t)kN * 4);
  float*  fdst  = (float*)alloc((size_t)kN * 4);
  float*  eatt  = (float*)alloc((size_t)kE * 4);
  float*  denom = (float*)alloc((size_t)kN * 4);
  float*  num   = (float*)alloc((size_t)kN * kD2 * 4);
  bf16_t* midbf = (bf16_t*)alloc((size_t)kN * kD1 * 2);   // layer-1 output (bf16)
  float*  twX   = (float*)alloc((size_t)kN * kD2 * 4);
  float*  tuX   = (float*)alloc((size_t)kN * kD2 * 4);
  bf16_t* twXbf = (bf16_t*)alloc((size_t)kN * kD2 * 2);
  bf16_t* tuXbf = (bf16_t*)alloc((size_t)kN * kD2 * 2);
  float*  scal  = (float*)alloc(4 * 4);  // [sum_tw, sum_tu, att0, att1]

  auto gemm = [&](const bf16_t* A, const bf16_t* B, float* C, int M, int Nn, int K,
                  int lda, int ldb, int ldc) {
    int waves = (M >> 4) * (Nn >> 6);   // 16x64 strip per wave
    wmma_gemm_kernel<<<cdiv((long long)waves * 32, 256), 256, 0, stream>>>(
        A, B, C, M, Nn, K, lda, ldb, ldc);
  };

  // one SpGAT view: Xbf (bf16 [N,Fp]) + edges -> outX (f32 [N,128]) + outXbf
  auto run_view = [&](const int* edges,
                      const float* W1, const float* a1,
                      const float* W2, const float* a2,
                      float* outX, bf16_t* outXbf) {
    const int* row = edges;
    const int* col = edges + kE;

    // ---- layer 1: d = 64, K = 300 (pad 320) ----
    conv_wt_kernel<<<cdiv(kD1 * kFp, 256), 256, 0, stream>>>(W1, Wt, kF, kD1, kFp);
    gemm(Xbf, Wt, h, kN, kD1, kFp, kFp, kFp, kD1);
    node_att_kernel<<<cdiv((long long)kN * 32, 256), 256, 0, stream>>>(h, a1, fsrc, fdst, kN, kD1);
    fill_f32_kernel<<<cdiv(kN, 256), 256, 0, stream>>>(denom, 0.f, kN);
    fill_f32_kernel<<<cdiv(kN * kD1, 256), 256, 0, stream>>>(num, 0.f, kN * kD1);
    edge_att_kernel<<<cdiv(kE, 256), 256, 0, stream>>>(row, col, fsrc, fdst, eatt, denom, kE);
    edge_agg_kernel<<<cdiv((long long)kE * kD1, 256), 256, 0, stream>>>(
        row, col, eatt, h, num, (long long)kE * kD1, kD1, 6);
    normalize_kernel<<<cdiv(kN * kD1, 256), 256, 0, stream>>>(
        num, denom, nullptr, midbf, kN * kD1, 6, 0);

    // ---- layer 2: d = 128, K = 64 ----
    conv_wt_kernel<<<cdiv(kD2 * kD1, 256), 256, 0, stream>>>(W2, Wt, kD1, kD2, kD1);
    gemm(midbf, Wt, h, kN, kD2, kD1, kD1, kD1, kD2);
    node_att_kernel<<<cdiv((long long)kN * 32, 256), 256, 0, stream>>>(h, a2, fsrc, fdst, kN, kD2);
    fill_f32_kernel<<<cdiv(kN, 256), 256, 0, stream>>>(denom, 0.f, kN);
    fill_f32_kernel<<<cdiv(kN * kD2, 256), 256, 0, stream>>>(num, 0.f, kN * kD2);
    edge_att_kernel<<<cdiv(kE, 256), 256, 0, stream>>>(row, col, fsrc, fdst, eatt, denom, kE);
    edge_agg_kernel<<<cdiv((long long)kE * kD2, 256), 256, 0, stream>>>(
        row, col, eatt, h, num, (long long)kE * kD2, kD2, 7);
    normalize_kernel<<<cdiv(kN * kD2, 256), 256, 0, stream>>>(
        num, denom, outX, outXbf, kN * kD2, 7, 1);
  };

  // tweet view
  tweet_feat_kernel<<<kN, 128, 0, stream>>>(features_index, word_emb, Xbf, kF, kFp);
  run_view(tw_edges, tw_W1, tw_a1, tw_W2, tw_a2, twX, twXbf);

  // user view (reuse Xbf)
  conv_x_kernel<<<cdiv(kN * kFp, 256), 256, 0, stream>>>(user_emb, Xbf, kN, kF, kFp);
  run_view(ut_edges, tu_W1, tu_a1, tu_W2, tu_a2, tuX, tuXbf);

  // ---- attention-weighted fusion ----
  fill_f32_kernel<<<1, 4, 0, stream>>>(scal, 0.f, 4);
  conv_wt_kernel<<<cdiv(kD2 * kD2, 256), 256, 0, stream>>>(weight_W, Wt, kD2, kD2, kD2);
  gemm(twXbf, Wt, h, kN, kD2, kD2, kD2, kD2, kD2);
  att_reduce_kernel<<<cdiv(kN, 8), 256, 0, stream>>>(h, weight_proj, scal + 0, kN, kD2);
  gemm(tuXbf, Wt, h, kN, kD2, kD2, kD2, kD2, kD2);
  att_reduce_kernel<<<cdiv(kN, 8), 256, 0, stream>>>(h, weight_proj, scal + 1, kN, kD2);
  att_softmax_kernel<<<1, 1, 0, stream>>>(scal, scal + 2, 1.f / (float)kN);

  // ---- head: gather + linear + log_softmax ----
  final_out_kernel<<<cdiv((long long)kB * 32, 256), 256, 0, stream>>>(
      twX, tuX, tw_gidx, ut_gidx, scal + 2, out_W, out_b, out, kB, kD2);
}
